// GNNLayer_15324443312566
// MI455X (gfx1250) — matrix operand order repaired and checked by reading, not compiled, not run on hardware
//
#include <hip/hip_runtime.h>

#define N_NODES   100000
#define N_EDGES   1600000
#define IN_F      256
#define OUT_F     64
#define ROW_BLOCKS (N_NODES / 16)   // 6250 exactly

typedef __attribute__((ext_vector_type(16))) __bf16 v16bf;
typedef __attribute__((ext_vector_type(8)))  float  v8f;

// Straight-line tanh: no exec-mask branches, uses v_exp_f32 + v_rcp_f32
// (TRANS pipe, co-issues with matrix ops). Exact at both saturation limits.
__device__ __forceinline__ float fast_tanhf(float x) {
    float e = __expf(2.0f * x);                 // v_exp_f32 (scaled)
    return 1.0f - 2.0f * __builtin_amdgcn_rcpf(e + 1.0f);
}

// ---------------------------------------------------------------------------
// Kernel 1: support = tanh(features @ weight), bf16 WMMA with fp32 accumulate.
// One wave computes a 16x64 output block (4 tiles of 16x16), K swept in 8
// chunks of 32. Weight is pre-swizzled into LDS in WMMA B-fragment order so
// each lane fetches its 16-element fragment as one contiguous 32B LDS read.
// All 4 B fragments of a chunk are loaded before the 4 WMMAs so the DS loads
// pipeline instead of serializing (wait dscnt 2/1/0 instead of 0/0/0).
// __launch_bounds__(256,4) caps the wave at 256 VGPRs (no vgpr-msb traffic,
// >=4 waves/SIMD to hide the HBM latency of the A stream).
// ---------------------------------------------------------------------------
__global__ __launch_bounds__(256, 4) void gnn_gemm_tanh_kernel(
    const float* __restrict__ features,
    const float* __restrict__ weight,
    const int*   __restrict__ active,
    float*       __restrict__ support)
{
    // B fragments: slot s = ((c*4 + t)*32 + lane), 16 bf16 each -> 32 KB
    __shared__ alignas(32) __bf16 ldsB[1024 * 16];

    const int tid = threadIdx.x;

    // Cooperative fill: B-fragment layout for V_WMMA_F32_16X16X32_BF16.
    // For B (32x16, KxN): lanes 0-15 hold K=0..7,16..23, lanes 16-31
    // hold K=8..15,24..31 (matching the A fragment order) at col n.
    for (int s = tid; s < 1024; s += 256) {
        const int l  = s & 31;          // lane this slot feeds
        const int t  = (s >> 5) & 3;    // N tile (0..3)
        const int c  = s >> 7;          // K chunk (0..7)
        const int n  = (l & 15) + t * 16;
        const int kb = c * 32 + (l >> 4) * 8;
        __bf16* dst = &ldsB[s * 16];
#pragma unroll
        for (int i = 0; i < 16; ++i)
            dst[i] = (__bf16)weight[(kb + i + (i >> 3) * 8) * OUT_F + n];
    }
    __syncthreads();

    const int lane   = tid & 31;
    const int h      = lane >> 4;   // lane half
    const int r      = lane & 15;   // row within 16-row block (A), col (B/C/D)
    const int wave   = (blockIdx.x * 256 + tid) >> 5;
    const int nwaves = (gridDim.x * 256) >> 5;
    const int act    = active[0];

    for (int rb = wave; rb < ROW_BLOCKS; rb += nwaves) {
        const float* __restrict__ arow =
            features + (size_t)(rb * 16 + r) * IN_F;

        v8f acc[4] = {};

#pragma unroll 2
        for (int c = 0; c < 8; ++c) {
            // A fragment (16x32 bf16): lane half h covers K = kb..kb+7 and
            // K = kb+16..kb+23 where kb = 32c + 8h (ISA 16-bit A layout).
            const int kb = c * 32 + h * 8;
            const float4 f0 = *(const float4*)(arow + kb);
            const float4 f1 = *(const float4*)(arow + kb + 4);
            const float4 f2 = *(const float4*)(arow + kb + 16);
            const float4 f3 = *(const float4*)(arow + kb + 20);

            v16bf a;
            a[0]  = (__bf16)f0.x; a[1]  = (__bf16)f0.y;
            a[2]  = (__bf16)f0.z; a[3]  = (__bf16)f0.w;
            a[4]  = (__bf16)f1.x; a[5]  = (__bf16)f1.y;
            a[6]  = (__bf16)f1.z; a[7]  = (__bf16)f1.w;
            a[8]  = (__bf16)f2.x; a[9]  = (__bf16)f2.y;
            a[10] = (__bf16)f2.z; a[11] = (__bf16)f2.w;
            a[12] = (__bf16)f3.x; a[13] = (__bf16)f3.y;
            a[14] = (__bf16)f3.z; a[15] = (__bf16)f3.w;

            // Load all 4 B fragments first so ds_load_b128 pairs pipeline.
            v16bf bfrag[4];
#pragma unroll
            for (int t = 0; t < 4; ++t)
                bfrag[t] =
                    *(const v16bf*)&ldsB[(((c << 2) | t) * 32 + lane) * 16];

#pragma unroll
            for (int t = 0; t < 4; ++t)
                acc[t] = __builtin_amdgcn_wmma_f32_16x16x32_bf16(
                    false, a, false, bfrag[t], (short)0, acc[t], false, false);
        }

        // Epilogue: tanh + store. C/D layout: VGPR v -> M = v + 8h, N = r.
#pragma unroll
        for (int t = 0; t < 4; ++t) {
#pragma unroll
            for (int v = 0; v < 8; ++v) {
                float val = acc[t][v];
                if (act) val = fast_tanhf(val);
                support[(size_t)(rb * 16 + v + 8 * h) * OUT_F + t * 16 + r] = val;
            }
        }
    }
}

// ---------------------------------------------------------------------------
// Kernel 2: y += A_sparse @ x  (rows sorted). One wave per contiguous edge
// chunk; each lane owns a float2 of the 64-wide row (256B coalesced gather).
// Register accumulation across equal-row runs; atomics only on row change.
// ---------------------------------------------------------------------------
__global__ __launch_bounds__(256) void gnn_spmm_kernel(
    const int*   __restrict__ rows,
    const int*   __restrict__ cols,
    const float* __restrict__ vals,
    const float* __restrict__ x,
    float*       __restrict__ y,
    int nedges)
{
    const int lane = threadIdx.x & 31;
    const int g    = (blockIdx.x * blockDim.x + threadIdx.x) >> 5;
    const int G    = (gridDim.x * blockDim.x) >> 5;
    const int chunk = (nedges + G - 1) / G;
    const int e0 = g * chunk;
    int e1 = e0 + chunk;
    if (e1 > nedges) e1 = nedges;
    if (e0 >= e1) return;

    float accx = 0.0f, accy = 0.0f;
    int prev = rows[e0];

    for (int e = e0; e < e1; ++e) {
        const int   rr = rows[e];
        const int   cc = cols[e];
        const float v  = vals[e];
        const float2 xv = ((const float2*)(x + (size_t)cc * OUT_F))[lane];
        if (rr != prev) {
            float* dst = y + (size_t)prev * OUT_F + lane * 2;
            __hip_atomic_fetch_add(dst + 0, accx, __ATOMIC_RELAXED,
                                   __HIP_MEMORY_SCOPE_AGENT);
            __hip_atomic_fetch_add(dst + 1, accy, __ATOMIC_RELAXED,
                                   __HIP_MEMORY_SCOPE_AGENT);
            accx = 0.0f; accy = 0.0f;
            prev = rr;
        }
        accx = fmaf(v, xv.x, accx);
        accy = fmaf(v, xv.y, accy);
    }
    float* dst = y + (size_t)prev * OUT_F + lane * 2;
    __hip_atomic_fetch_add(dst + 0, accx, __ATOMIC_RELAXED,
                           __HIP_MEMORY_SCOPE_AGENT);
    __hip_atomic_fetch_add(dst + 1, accy, __ATOMIC_RELAXED,
                           __HIP_MEMORY_SCOPE_AGENT);
}

// ---------------------------------------------------------------------------
// Launch: d_out = [output (N*64) | az (N*64)]. The az half doubles as scratch
// for `support` until SpMM-1 finishes, then is cleared and refilled as az.
// ---------------------------------------------------------------------------
extern "C" void kernel_launch(void* const* d_in, const int* in_sizes, int n_in,
                              void* d_out, int out_size, void* d_ws, size_t ws_size,
                              hipStream_t stream)
{
    const float* features = (const float*)d_in[0];
    const float* weight   = (const float*)d_in[1];
    const int*   adj_rows = (const int*)d_in[2];
    const int*   adj_cols = (const int*)d_in[3];
    const float* adj_vals = (const float*)d_in[4];
    const int*   active   = (const int*)d_in[5];

    float* out_first  = (float*)d_out;                                // output
    float* out_second = out_first + (size_t)N_NODES * OUT_F;          // az
    const size_t half_bytes = (size_t)N_NODES * OUT_F * sizeof(float);

    // 1) zero the `output` accumulation region
    hipMemsetAsync(out_first, 0, half_bytes, stream);

    // 2) support = tanh(features @ weight)  -> stored in az half (temp)
    gnn_gemm_tanh_kernel<<<392, 256, 0, stream>>>(
        features, weight, active, out_second);

    // 3) output = adj @ support
    gnn_spmm_kernel<<<1024, 256, 0, stream>>>(
        adj_rows, adj_cols, adj_vals, out_second, out_first, N_EDGES);

    // 4) clear az half, then az = adj @ output
    hipMemsetAsync(out_second, 0, half_bytes, stream);
    gnn_spmm_kernel<<<1024, 256, 0, stream>>>(
        adj_rows, adj_cols, adj_vals, out_first, out_second, N_EDGES);
}
